// TransformerBlockQuantum_65481071405168
// MI455X (gfx1250) — matrix operand order repaired and checked by reading, not compiled, not run on hardware
//
#include <hip/hip_runtime.h>
#include <cstdint>

typedef __attribute__((ext_vector_type(2)))  float    v2f;
typedef __attribute__((ext_vector_type(8)))  float    v8f;
typedef __attribute__((ext_vector_type(2)))  _Float16 v2h;
typedef __attribute__((ext_vector_type(16))) _Float16 v16h;

#define B_ 16
#define S_ 2048
#define E_ 8
#define F_ 512
#define H_ 2

// ---------------------------------------------------------------------------
// Phase 1: quantum head  z[b,s,0:8]  (cumprod of cos), one thread per (b,s)
// ---------------------------------------------------------------------------
__global__ __launch_bounds__(256) void qz_kernel(const float* __restrict__ x,
                                                 const float* __restrict__ th,
                                                 float* __restrict__ z) {
  int i = blockIdx.x * blockDim.x + threadIdx.x;       // row id 0..B*S-1
  const float* xr = x + (long)i * 8;
  float c[8];
#pragma unroll
  for (int e = 0; e < 8; ++e) c[e] = __cosf(xr[e] + th[e]);
  float o[8];
  float p = c[1];
#pragma unroll
  for (int e = 2; e < 8; ++e) p *= c[e];
  o[0] = p;                                            // z0 = prod c1..c7
  float run = c[0];
#pragma unroll
  for (int e = 1; e < 8; ++e) { run *= c[e]; o[e] = run; }  // cumprod
  float4* zo = (float4*)(z + (long)i * 8);
  zo[0] = make_float4(o[0], o[1], o[2], o[3]);
  zo[1] = make_float4(o[4], o[5], o[6], o[7]);
}

// ---------------------------------------------------------------------------
// Phase 1b: build f16 B-panels for the P*V WMMA.
// Per (b,h,32-key block): B matrix (K=32 keys x N=16), cols 0..3 = V dims,
// col 4 = 1.0 (softmax denominator), rest 0.  Stored in the exact per-lane
// VGPR layout of V_WMMA_F32_16X16X32_F16's B operand:
//   lane l: N = l&15, VGPR v holds K=8*hs+2v,+1 (v<4), K=16+8*hs+2(v-4),+1 (v>=4).
// One wave per tile; each lane writes its 32 contiguous bytes.
// ---------------------------------------------------------------------------
__global__ __launch_bounds__(256) void pan_kernel(const float* __restrict__ z,
                                                  unsigned* __restrict__ pan) {
  int t    = (blockIdx.x * blockDim.x + threadIdx.x) >> 5;   // tile 0..2047
  int lane = threadIdx.x & 31;
  int p  = t & 63;
  int bh = t >> 6;
  int h  = bh & 1;
  int b  = bh >> 1;
  int n  = lane & 15;
  int hs = lane >> 4;
  unsigned o[8];
#pragma unroll
  for (int v = 0; v < 8; ++v) {
    int key = p * 32 + hs * 8 + (v >> 2) * 8 + 2 * v;
    float lo, hi;
    if (n < 4) {
      lo = z[((long)(b * S_ + key    )) * 8 + h * 4 + n];
      hi = z[((long)(b * S_ + key + 1)) * 8 + h * 4 + n];
    } else if (n == 4) { lo = 1.0f; hi = 1.0f; }
    else               { lo = 0.0f; hi = 0.0f; }
    union { _Float16 hf[2]; unsigned u; } pk;
    pk.hf[0] = (_Float16)lo;     // low half = lower K
    pk.hf[1] = (_Float16)hi;
    o[v] = pk.u;
  }
  uint4* dst = (uint4*)(pan + (long)t * 256 + lane * 8);
  dst[0] = make_uint4(o[0], o[1], o[2], o[3]);
  dst[1] = make_uint4(o[4], o[5], o[6], o[7]);
}

// ---------------------------------------------------------------------------
// Phase 2: flash attention. One wave per (b, h, 16-query tile).
// Scores are computed TRANSPOSED: D = S^T = K_tile(16x4) x Q_tile^T(4x16) via
// V_WMMA_F32_16X16X4_F32, with 0.5*log2(e) folded into the loop-invariant
// query operand.  In S^T's D-layout, lane l holds query N=l&15 and keys
// r+hs*8 in its own registers -- exactly the per-lane halves the f16 A-operand
// of V_WMMA_F32_16X16X32_F16 wants (hs=0: K=0..7/16..23, hs=1: K=8..15/24..31).
// So exp2 + v_cvt_pk_rtz feed the P*V WMMA with no LDS and no lane shuffles.
// Softmax denominator rides along as a ones-column in the V panel (col 4).
// exp without max-subtraction is safe: |z_i|<=1 => |q.k/2| <= 2.
// ---------------------------------------------------------------------------
__global__ __launch_bounds__(256) void attn_kernel(const float* __restrict__ z,
                                                   const unsigned* __restrict__ pan,
                                                   float* __restrict__ attn) {
  int wv   = threadIdx.x >> 5;
  int lane = threadIdx.x & 31;
  int gw   = blockIdx.x * 8 + wv;                      // 0..4095
  int qt = gw & 127;
  int bh = gw >> 7;
  int h  = bh & 1;
  int b  = bh >> 1;
  int m  = lane & 15;
  int hs = lane >> 4;
  int col = h * 4 + hs * 2;

  // B operand (query tile^T, f32 4x16 B-layout): lane: N=m, K=hs*2+{0,1}.
  const float sc = 0.72134752044448170f;               // 0.5 * log2(e)
  const float* zq = z + (((long)b * S_ + qt * 16 + m) * 8 + col);
  v2f bq = { zq[0] * sc, zq[1] * sc };

  v8f acc   = {};
  v8f czero = {};
  const float* zkb = z + ((long)b * S_) * 8 + col;
  const uint4* panw = (const uint4*)pan + (long)bh * 64 * 64;  // 64 tiles x 64 uint4

  for (int p = 0; p < 64; ++p) {
    union { v2h h2[8]; v16h v; } Am;
#pragma unroll
    for (int t = 0; t < 2; ++t) {
      int k0 = p * 32 + t * 16;
      // A operand (key tile, f32 16x4 A-layout): lane: M=m -> key k0+m.
      const float* zk = zkb + (long)(k0 + m) * 8;
      v2f ak = { zk[0], zk[1] };
      v8f s = __builtin_amdgcn_wmma_f32_16x16x4_f32(false, ak, false, bq,
                                                    (short)0, czero, false, false);
      // s[r] = scaled score(key r+hs*8, query m).  exp2 + pack pairs:
      // VGPR v of the f16 A operand = keys (2v,2v+1)+hs*8 of tile t.
#pragma unroll
      for (int v = 0; v < 4; ++v) {
        float p0 = __builtin_amdgcn_exp2f(s[2 * v]);
        float p1 = __builtin_amdgcn_exp2f(s[2 * v + 1]);
        Am.h2[t * 4 + v] = __builtin_bit_cast(v2h, __builtin_amdgcn_cvt_pkrtz(p0, p1));
      }
    }
    union { uint4 u[2]; v16h v; } Bm;
    const uint4* pp = panw + p * 64 + lane * 2;
    Bm.u[0] = pp[0];
    Bm.u[1] = pp[1];
    acc = __builtin_amdgcn_wmma_f32_16x16x32_f16(false, Am.v, false, Bm.v,
                                                 (short)0, acc, false, false);
  }
  // acc: lane l = col N (0..3 = sum P*v, 4 = sum P), rows M = r+hs*8 = query.
  int srcl = 4 + (hs << 4);
#pragma unroll
  for (int r = 0; r < 8; ++r) {
    float den = __shfl(acc[r], srcl, 32);
    float o = acc[r] / den;
    if (m < 4) {
      attn[((long)b * S_ + qt * 16 + r + hs * 8) * 8 + h * 4 + m] = o;
    }
  }
}

// ---------------------------------------------------------------------------
// Phase 3: out-proj + residual + LN1 + quantum FFN + MLP(8->512->8) + LN2.
// One thread per (b,s) row; w1 / w2^T staged in LDS (uniform broadcast reads).
// ---------------------------------------------------------------------------
__global__ __launch_bounds__(256) void ffn_kernel(
    const float* __restrict__ x, const float* __restrict__ attn,
    const float* __restrict__ w_out, const float* __restrict__ th_ry,
    const float* __restrict__ w1, const float* __restrict__ b1,
    const float* __restrict__ w2, const float* __restrict__ b2,
    const float* __restrict__ g1, const float* __restrict__ be1,
    const float* __restrict__ g2, const float* __restrict__ be2,
    float* __restrict__ out) {
  __shared__ float w1s[F_ * E_];
  __shared__ float w2t[F_ * E_];
  __shared__ float b1s[F_];
  for (int i = threadIdx.x; i < F_ * E_; i += 256) w1s[i] = w1[i];
  for (int i = threadIdx.x; i < F_ * E_; i += 256) {
    int f = i >> 3, e = i & 7;
    w2t[i] = w2[e * F_ + f];                           // transpose: row f contiguous
  }
  for (int i = threadIdx.x; i < F_; i += 256) b1s[i] = b1[i];
  __syncthreads();

  long row = (long)blockIdx.x * 256 + threadIdx.x;
  const float* xr = x + row * 8;
  const float* ar = attn + row * 8;
  float at[8];
#pragma unroll
  for (int e = 0; e < 8; ++e) at[e] = ar[e];
  float t[8]; float mu = 0.f;
#pragma unroll
  for (int e = 0; e < 8; ++e) {
    float s = 0.f;
#pragma unroll
    for (int j = 0; j < 8; ++j) s += at[j] * w_out[e * 8 + j];
    t[e] = xr[e] + s; mu += t[e];
  }
  mu *= 0.125f;
  float var = 0.f;
#pragma unroll
  for (int e = 0; e < 8; ++e) { float d = t[e] - mu; var += d * d; }
  var *= 0.125f;
  float inv = rsqrtf(var + 1e-5f);
  float x1[8], zf[8];
#pragma unroll
  for (int e = 0; e < 8; ++e) {
    x1[e] = (t[e] - mu) * inv * g1[e] + be1[e];
    zf[e] = __cosf(th_ry[e]) * __cosf(x1[e]);
  }
  float fa[8] = {0, 0, 0, 0, 0, 0, 0, 0};
  for (int f = 0; f < F_; ++f) {
    const float4* wr = (const float4*)&w1s[f * 8];
    float4 wa = wr[0], wb = wr[1];
    float hv = b1s[f]
      + zf[0] * wa.x + zf[1] * wa.y + zf[2] * wa.z + zf[3] * wa.w
      + zf[4] * wb.x + zf[5] * wb.y + zf[6] * wb.z + zf[7] * wb.w;
    hv = fmaxf(hv, 0.0f);
    const float4* ur = (const float4*)&w2t[f * 8];
    float4 ua = ur[0], ub = ur[1];
    fa[0] += hv * ua.x; fa[1] += hv * ua.y; fa[2] += hv * ua.z; fa[3] += hv * ua.w;
    fa[4] += hv * ub.x; fa[5] += hv * ub.y; fa[6] += hv * ub.z; fa[7] += hv * ub.w;
  }
  float t2[8]; mu = 0.f;
#pragma unroll
  for (int e = 0; e < 8; ++e) { t2[e] = x1[e] + fa[e] + b2[e]; mu += t2[e]; }
  mu *= 0.125f;
  var = 0.f;
#pragma unroll
  for (int e = 0; e < 8; ++e) { float d = t2[e] - mu; var += d * d; }
  var *= 0.125f;
  inv = rsqrtf(var + 1e-5f);
  float o[8];
#pragma unroll
  for (int e = 0; e < 8; ++e) o[e] = (t2[e] - mu) * inv * g2[e] + be2[e];
  float4* op = (float4*)(out + row * 8);
  op[0] = make_float4(o[0], o[1], o[2], o[3]);
  op[1] = make_float4(o[4], o[5], o[6], o[7]);
}

// ---------------------------------------------------------------------------
extern "C" void kernel_launch(void* const* d_in, const int* in_sizes, int n_in,
                              void* d_out, int out_size, void* d_ws, size_t ws_size,
                              hipStream_t stream) {
  const float* x     = (const float*)d_in[0];
  const float* th_rx = (const float*)d_in[1];
  const float* w_out = (const float*)d_in[2];
  const float* th_ry = (const float*)d_in[3];
  const float* w1    = (const float*)d_in[4];
  const float* b1    = (const float*)d_in[5];
  const float* w2    = (const float*)d_in[6];
  const float* b2    = (const float*)d_in[7];
  const float* g1    = (const float*)d_in[8];
  const float* be1   = (const float*)d_in[9];
  const float* g2    = (const float*)d_in[10];
  const float* be2   = (const float*)d_in[11];
  float* out = (float*)d_out;

  char* ws = (char*)d_ws;
  float*    z    = (float*)(ws);                 // 1 MB : quantum-head features
  unsigned* pan  = (unsigned*)(ws + (1u << 20)); // 2 MB : f16 V|ones B-panels
  float*    attn = (float*)(ws + (3u << 20));    // 1 MB : attention output

  qz_kernel  <<<128, 256, 0, stream>>>(x, th_rx, z);
  pan_kernel <<<256, 256, 0, stream>>>(z, pan);
  attn_kernel<<<512, 256, 0, stream>>>(z, pan, attn);
  ffn_kernel <<<128, 256, 0, stream>>>(x, attn, w_out, th_ry, w1, b1, w2, b2,
                                       g1, be1, g2, be2, out);
}
